// Compact_Triplet_MMD_Entropy_78889959293572
// MI455X (gfx1250) — matrix operand order repaired and checked by reading, not compiled, not run on hardware
//
#include <hip/hip_runtime.h>
#include <hip/hip_bf16.h>

// ---------------------------------------------------------------------------
// Problem constants (from reference)
// ---------------------------------------------------------------------------
#define LSEQ   512          // MAXLEN
#define FEATN  64           // FEAT
#define NWK    4            // NW
#define STEPK  16           // STEP = 1 + 5 + 5 + 5
#define NPAIR  60           // NW * M_OTH
#define BIGF   1e10f
#define GAMF   5.0f
#define INVG   0.2f         // 1 / GAMMA

typedef __attribute__((ext_vector_type(16))) _Float16 v16h;
typedef __attribute__((ext_vector_type(8)))  float    v8f;

// ---------------------------------------------------------------------------
// Tensor Data Mover staging (gfx1250). Builds a D# descriptor (ISA 08 §8.3/8.4)
// for a 2D tile of f32: tile_d0 contiguous elements per row, tile_d1 rows at
// stride0 elements. Falls back to a cooperative copy if the builtin is absent.
// ---------------------------------------------------------------------------
#if defined(__HIP_DEVICE_COMPILE__) && \
    __has_builtin(__builtin_amdgcn_tensor_load_to_lds) && \
    __has_builtin(__builtin_amdgcn_s_wait_tensorcnt)
#define USE_TDM 1
#else
#define USE_TDM 0
#endif

#if USE_TDM
typedef __attribute__((ext_vector_type(4))) unsigned int v4u;
typedef __attribute__((ext_vector_type(4))) int          v4i;
typedef __attribute__((ext_vector_type(8))) int          v8i;

__device__ __forceinline__ void tdm_issue_2d(const void* gsrc, unsigned lds_off,
                                             unsigned tile_d0, unsigned tile_d1,
                                             unsigned long long stride0)
{
    const unsigned long long ga = (unsigned long long)(uintptr_t)gsrc;
    v4u g0;
    g0[0] = 1u;                                                  // count=1, user D#
    g0[1] = lds_off;                                             // lds_addr (bytes)
    g0[2] = (unsigned)(ga & 0xFFFFFFFFull);                      // global_addr[31:0]
    g0[3] = (unsigned)((ga >> 32) & 0x1FFFFFFull) | (2u << 30);  // [56:32] | type=2
    v8i g1;
    g1[0] = (int)(2u << 16);                                     // data_size=4B, mask=0
    g1[1] = (int)((tile_d0 & 0xFFFFu) << 16);                    // tensor_dim0 lo16
    g1[2] = (int)(((tile_d0 >> 16) & 0xFFFFu) |
                  ((tile_d1 & 0xFFFFu) << 16));                  // td0 hi | tensor_dim1 lo
    g1[3] = (int)(((tile_d1 >> 16) & 0xFFFFu) |
                  ((tile_d0 & 0xFFFFu) << 16));                  // td1 hi | tile_dim0
    g1[4] = (int)(tile_d1 & 0xFFFFu);                            // tile_dim1 (tile_dim2=0)
    g1[5] = (int)(stride0 & 0xFFFFFFFFull);                      // tensor_dim0_stride lo
    g1[6] = (int)((stride0 >> 32) & 0xFFFFull);                  // stride hi | dim1_stride=0
    g1[7] = 0;
    const v4i gz = {0, 0, 0, 0};
#if defined(__clang_major__) && (__clang_major__ >= 23)
    const v8i gz8 = {0, 0, 0, 0, 0, 0, 0, 0};
    __builtin_amdgcn_tensor_load_to_lds(g0, g1, gz, gz, gz8, 0);
#else
    __builtin_amdgcn_tensor_load_to_lds(g0, g1, gz, gz, 0);
#endif
}
#endif

// issue (wave 0 only under TDM) — caller must __syncthreads() after stage_wait
__device__ __forceinline__ void stage_2d(const float* g, float* dst,
                                         int d0log2, int d0, int d1,
                                         long long stride, int tid, int nthr)
{
#if USE_TDM
    (void)d0log2; (void)nthr;
    if (tid < 32)
        tdm_issue_2d(g, (unsigned)(uintptr_t)dst,
                     (unsigned)d0, (unsigned)d1, (unsigned long long)stride);
#else
    for (int e = tid; e < d0 * d1; e += nthr)
        dst[e] = g[(long long)(e >> d0log2) * stride + (e & (d0 - 1))];
#endif
}

__device__ __forceinline__ void stage_wait(int tid)
{
#if USE_TDM
    if (tid < 32) __builtin_amdgcn_s_wait_tensorcnt(0);
#else
    (void)tid;
#endif
}

// ---------------------------------------------------------------------------
// Kernel 1: fused pairwise-sqdist (WMMA f16 tiles, on the fly) + soft-DTW
// block-antidiagonal wavefront. One workgroup (16 waves) per pair. X and Y
// staged HBM -> LDS (TDM) -> f16; D never touches HBM.
// ---------------------------------------------------------------------------
__global__ __launch_bounds__(512) void dtw_pair_kernel(
    const float* __restrict__ data,
    const int*   __restrict__ lens,
    float*       __restrict__ r_out)
{
    extern __shared__ __align__(16) char smem[];
    _Float16* Xh    = (_Float16*)smem;                 // 32768 halves (64 KB)
    _Float16* Yh    = Xh + LSEQ * FEATN;               // 32768 halves (64 KB)
    float*    nx    = (float*)(Yh + LSEQ * FEATN);     // 512
    float*    ny    = nx + LSEQ;                       // 512
    float*    Hrow  = ny + LSEQ;                       // 516 (use 0..512)
    float*    Vcol  = Hrow + 516;                      // 516 (use 0..512)
    float*    Cor   = Vcol + 516;                      // 128 (3 slots x 33 used)
    float*    dts   = Cor + 128;                       // 16 waves * 256 f32
    float*    stage = dts + 16 * 256;                  // 16384 f32 (64 KB), 16B-aligned

    const int pair = blockIdx.x;                       // 0..59
    const int w    = pair / 15;
    const int m    = pair % 15;
    const int aidx = w * STEPK;
    const int oidx = aidx + 1 + m;
    const float* X = data + (size_t)aidx * (LSEQ * FEATN);
    const float* Y = data + (size_t)oidx * (LSEQ * FEATN);
    const int la = lens[aidx];
    const int lb = lens[oidx];

    const int tid = threadIdx.x;

    // ---- stage X, Y via TDM in 128-row chunks, convert to f16 in LDS ----
    for (int ch = 0; ch < 4; ++ch) {
        stage_2d(X + ch * 8192, stage,        13, 8192, 1, 8192, tid, 512);
        stage_2d(Y + ch * 8192, stage + 8192, 13, 8192, 1, 8192, tid, 512);
        stage_wait(tid);
        __syncthreads();
        for (int e = tid; e < 8192; e += 512) {
            Xh[ch * 8192 + e] = (_Float16)stage[e];
            Yh[ch * 8192 + e] = (_Float16)stage[8192 + e];
        }
        __syncthreads();
    }

    // ---- row squared-norms (from staged f16) ----
    for (int rr = tid; rr < LSEQ; rr += 512) {
        float sx = 0.f, sy = 0.f;
        for (int k = 0; k < FEATN; ++k) {
            float xv = (float)Xh[rr * FEATN + k];
            float yv = (float)Yh[rr * FEATN + k];
            sx += xv * xv;
            sy += yv * yv;
        }
        nx[rr] = sx;
        ny[rr] = sy;
    }
    // ---- boundary init: r[0][j>0]=BIG, r[i>0][0]=BIG, r[0][0]=0 ----
    for (int j = tid; j <= LSEQ; j += 512) { Hrow[j] = BIGF; Vcol[j] = BIGF; }
    for (int q = tid; q < 128; q += 512) Cor[q] = BIGF;
    if (tid == 0) Cor[1 * 33 + 0] = 0.0f;   // corner for tile (0,0) read at bd=0
    __syncthreads();

    const int wv   = tid >> 5;
    const int lane = tid & 31;
    const int r    = lane & 15;             // intra-tile row owned by this lane
    const bool lowhalf = (lane < 16);
    const bool isr0    = (r == 0);

    // 32x32 tiles of 16x16, 63 block-antidiagonals
    for (int bd = 0; bd < 63; ++bd) {
        const int lo    = (bd > 31) ? bd - 31 : 0;
        const int hi    = (bd < 31) ? bd : 31;
        const int count = hi - lo + 1;

        for (int tix = wv; tix < count; tix += 16) {
            const int bi = lo + tix;
            const int bj = bd - bi;
            const int i0 = bi * 16;
            const int j0 = bj * 16;

            // -------- WMMA: dot tile = X[i0..i0+15][:] * Y[j0..j0+15][:]^T --
            v8f acc = {};
            const int arow = i0 + (lane & 15);
            const int brow = j0 + (lane & 15);
            const int ak0  = (lane >> 4) ? 16 : 0;
            const int bk0  = (lane >> 4) ? 16 : 0;
            #pragma unroll
            for (int kk = 0; kk < 2; ++kk) {
                v16h a, b;
                #pragma unroll
                for (int t = 0; t < 8; ++t) {
                    a[t]     = Xh[arow * FEATN + kk * 32 + ak0 + t];
                    a[t + 8] = Xh[arow * FEATN + kk * 32 + ak0 + 8 + t];
                }
                #pragma unroll
                for (int t = 0; t < 16; ++t)
                    b[t] = Yh[brow * FEATN + kk * 32 + bk0 + t];
                acc = __builtin_amdgcn_wmma_f32_16x16x32_f16(
                        false, a, false, b, (short)0, acc, false, false);
            }
            // dump 16x16 f32 tile to per-wave LDS scratch (C/D layout)
            float* dt = dts + (wv << 8);
            {
                const int M0 = (lane >> 4) ? 8 : 0;
                const int n  = lane & 15;
                #pragma unroll
                for (int v = 0; v < 8; ++v) dt[(M0 + v) * 16 + n] = acc[v];
            }

            // -------- read tile boundaries --------
            const float htop   = Hrow[j0 + 1 + r];                  // r[i0][j0+1+r]
            const float vleft  = Vcol[i0 + 1 + r];                  // r[i0+1+r][j0]
            const float corner = Cor[((bd + 1) % 3) * 33 + bj];     // r[i0][j0]
            const float nxr    = nx[i0 + r];
            const float nyr    = ny[j0 + r];
            float hm1 = __shfl_up(htop, 1, 32);   // lane l: Htop[l-1]
            hm1 = isr0 ? corner : hm1;            // lane 0: corner

            // -------- intra-tile 31-step lane wavefront (branch-lean) ------
            float cur    = vleft;   // lane's r[i][j0] before column 0
            float pneigh = 0.0f;
            const int ig = i0 + 1 + r;

            for (int t = 0; t < 31; ++t) {
                const float neigh = __shfl_up(cur, 1, 32);
                const int c  = t - r;
                const int cc = c & 15;
                const float upt = __shfl(htop, cc, 32);
                const float dgt = __shfl(hm1,  cc, 32);
                const float nyc = __shfl(nyr,  cc, 32);
                if (c >= 0 && c < 16) {
                    const float up   = isr0 ? upt : neigh;
                    const float dg   = isr0 ? dgt : pneigh;
                    const float left = cur;
                    const float mn = fminf(up, fminf(dg, left));
                    const float e =
                        __expf((mn - up)   * INVG) +
                        __expf((mn - dg)   * INVG) +
                        __expf((mn - left) * INVG);
                    const float smin = mn - GAMF * __logf(e);
                    const float dot  = dt[(r << 4) + c];
                    const float dval = fmaxf(nxr + nyc - 2.0f * dot, 0.0f);
                    cur = dval + smin;
                    if (lowhalf) {
                        if (r == 15) Hrow[j0 + 1 + c] = cur;       // bottom edge
                        if (ig == la && (j0 + 1 + c) == lb) r_out[pair] = cur;
                    }
                }
                pneigh = neigh;
            }
            if (lowhalf) {
                Vcol[i0 + 1 + r] = cur;                            // right edge
                if (r == 15) Cor[(bd % 3) * 33 + (bj + 1)] = cur;  // corner fwd
                if (r == 0 && bj == 0) Cor[(bd % 3) * 33 + 0] = BIGF;
            }
        }
        __syncthreads();
    }
}

// ---------------------------------------------------------------------------
// Kernel 2: MMD per pair. Gram of total (22 x 32768) via WMMA f16 tiles with
// TDM-staged 11-row strided chunks, then single-thread epilogue.
// ---------------------------------------------------------------------------
__constant__ int c_mmd_pairs[9][2] = {
    {0,1},{0,2},{0,3},{1,2},{1,3},{2,1},{2,3},{3,1},{3,2}
};

__global__ __launch_bounds__(128) void mmd_kernel(
    const float* __restrict__ data,
    float*       __restrict__ mmds)
{
    __shared__ float    stageF[22 * 128];   // TDM landing zone (11 KB)
    __shared__ _Float16 Th[32][128];        // padded 22 -> 32 rows (8 KB)
    __shared__ float    gram[32][32];

    const int p    = blockIdx.x;            // 0..8
    const int wi   = c_mmd_pairs[p][0];
    const int wj   = c_mmd_pairs[p][1];
    const int tid  = threadIdx.x;
    const int lane = tid & 31;
    const int wv   = tid >> 5;
    const int rt   = wv >> 1;               // output row tile (0..1)
    const int ct   = wv & 1;                // output col tile (0..1)

    const float* srcI = data + (size_t)(wi * STEPK) * (LSEQ * FEATN);
    const float* srcJ = data + (size_t)(wj * STEPK) * (LSEQ * FEATN);

    // zero padded rows once
    for (int e = tid; e < 32 * 128; e += 128) {
        if ((e >> 7) >= 22) Th[e >> 7][e & 127] = (_Float16)0.0f;
    }

    v8f acc = {};
    for (int ch = 0; ch < (LSEQ * FEATN) / 128; ++ch) {   // 256 K-chunks of 128
        // 2D strided gather: 11 rows x 128 f32, row stride = 32768 elements
        stage_2d(srcI + ch * 128, stageF,            7, 128, 11, LSEQ * FEATN, tid, 128);
        stage_2d(srcJ + ch * 128, stageF + 11 * 128, 7, 128, 11, LSEQ * FEATN, tid, 128);
        stage_wait(tid);
        __syncthreads();
        for (int e = tid; e < 22 * 128; e += 128)
            Th[e >> 7][e & 127] = (_Float16)stageF[e];
        __syncthreads();

        const int arow = rt * 16 + (lane & 15);
        const int brow = ct * 16 + (lane & 15);
        const int ak0  = (lane >> 4) ? 16 : 0;
        const int bk0  = (lane >> 4) ? 16 : 0;
        #pragma unroll
        for (int kk = 0; kk < 4; ++kk) {
            v16h a, b;
            #pragma unroll
            for (int t = 0; t < 8; ++t) {
                a[t]     = Th[arow][kk * 32 + ak0 + t];
                a[t + 8] = Th[arow][kk * 32 + ak0 + 8 + t];
            }
            #pragma unroll
            for (int t = 0; t < 16; ++t) b[t] = Th[brow][kk * 32 + bk0 + t];
            acc = __builtin_amdgcn_wmma_f32_16x16x32_f16(
                    false, a, false, b, (short)0, acc, false, false);
        }
        __syncthreads();   // protect stageF before next chunk's TDM overwrite
    }
    // dump gram tiles
    {
        const int M0 = (lane >> 4) ? 8 : 0;
        const int n  = lane & 15;
        #pragma unroll
        for (int v = 0; v < 8; ++v)
            gram[rt * 16 + M0 + v][ct * 16 + n] = acc[v];
    }
    __syncthreads();

    if (tid == 0) {
        const int n = 22;
        float tot = 0.0f;
        for (int a = 0; a < n; ++a)
            for (int b = 0; b < n; ++b) {
                float l2 = gram[a][a] + gram[b][b] - 2.0f * gram[a][b];
                tot += fmaxf(l2, 0.0f);
            }
        float bw = tot / (float)(n * n - n);
        bw *= 0.25f;                         // / KMUL^(KNUM//2) = / 4
        float inv_bws[5];
        float cb = bw;
        #pragma unroll
        for (int k = 0; k < 5; ++k) { inv_bws[k] = 1.0f / cb; cb *= 2.0f; }

        float s = 0.0f;
        for (int a = 0; a < n; ++a)
            for (int b = 0; b < n; ++b) {
                float l2 = fmaxf(gram[a][a] + gram[b][b] - 2.0f * gram[a][b], 0.0f);
                float kv = 0.0f;
                #pragma unroll
                for (int k = 0; k < 5; ++k) kv += __expf(-l2 * inv_bws[k]);
                const bool ia = (a < 11), ib = (b < 11);
                s += (ia == ib) ? kv : -kv;
            }
        mmds[p] = s / 121.0f;                // mean over 11x11
    }
}

// ---------------------------------------------------------------------------
// Kernel 3: tiny epilogue — triplet / BCE / intra / inter over 4x15 dists,
// plus mmd max. Writes (loss, nz_r_sum) to d_out.
// ---------------------------------------------------------------------------
__global__ void combine_kernel(
    const float* __restrict__ r_out,
    const int*   __restrict__ lens,
    const float* __restrict__ mmds,
    float*       __restrict__ out)
{
    if (threadIdx.x != 0 || blockIdx.x != 0) return;

    float dists[NWK][15];
    for (int pr = 0; pr < NPAIR; ++pr) {
        const int w = pr / 15, m = pr % 15;
        const int la = lens[w * STEPK];
        const int lb = lens[w * STEPK + 1 + m];
        dists[w][m] = r_out[pr] / (float)(la + lb);
    }

    float total = 0.0f;
    int nzr_total = 0;
    for (int w = 0; w < NWK; ++w) {
        const float* dg = dists[w];          // [0..4]  good
        const float* dn = dists[w] + 5;      // [0..9]  first 5 f, next 5 random
        float lks = 0.0f, lkr = 0.0f;
        int nzs = 0, nzr = 0;
        for (int i = 0; i < 5; ++i)
            for (int j = 0; j < 5; ++j) {
                float vs = dg[i] + 1.0f - dn[j];
                if (vs > 0.0f) { lks += vs; ++nzs; }
                float vr = dg[i] + 1.5f - dn[5 + j];
                if (vr > 0.0f) { lkr += vr; ++nzr; }
            }
        float ca = 0.0f; for (int i = 0; i < 5; ++i) ca += dg[i]; ca *= 0.2f;
        float cbm = 0.0f; for (int i = 0; i < 4; ++i) cbm += dn[i]; cbm *= 0.25f;
        float intra = 0.0f; for (int i = 0; i < 5; ++i) intra += dg[i] - ca;
        float inter = fmaxf(1.0f - fabsf(ca - cbm), 0.0f);
        float lv = (lks + lkr) / (float)(nzs + nzr + 1);
        float bce = 0.0f;
        for (int j = 0; j < 5; ++j) {
            float x = dn[5 + j] - dg[0];
            bce += (x >= 0.0f) ? log1pf(expf(-x)) : (-x + log1pf(expf(x)));
        }
        bce *= 0.2f;
        total += bce + lv + intra * 0.01f + inter * 0.01f;
        nzr_total += nzr;
    }
    total *= 0.25f;                          // mean over NW

    float mx = 0.0f;                         // mmds padded with zeros in ref
    for (int p = 0; p < 9; ++p) mx = fmaxf(mx, mmds[p]);

    out[0] = total + mx * 0.1f;
    out[1] = (float)nzr_total;
}

// ---------------------------------------------------------------------------
extern "C" void kernel_launch(void* const* d_in, const int* in_sizes, int n_in,
                              void* d_out, int out_size, void* d_ws, size_t ws_size,
                              hipStream_t stream) {
    (void)in_sizes; (void)n_in; (void)out_size; (void)ws_size;
    const float* data = (const float*)d_in[0];
    const int*   lens = (const int*)d_in[1];

    float* r_out = (float*)d_ws;                       // 60 floats
    float* mmds  = (float*)((char*)d_ws + 256);        // 9 floats

    const size_t shmem =
        (size_t)2 * LSEQ * FEATN * sizeof(_Float16) +                  // Xh, Yh
        (size_t)(2 * LSEQ + 2 * 516 + 128 + 16 * 256 + 16384) * sizeof(float);

    hipLaunchKernelGGL(dtw_pair_kernel, dim3(NPAIR), dim3(512), shmem, stream,
                       data, lens, r_out);
    hipLaunchKernelGGL(mmd_kernel, dim3(9), dim3(128), 0, stream, data, mmds);
    hipLaunchKernelGGL(combine_kernel, dim3(1), dim3(64), 0, stream,
                       r_out, lens, mmds, (float*)d_out);
}
